// AbsoluteTopKSAE_73658689126867
// MI455X (gfx1250) — matrix operand order repaired and checked by reading, not compiled, not run on hardware
//
#include <hip/hip_runtime.h>
#include <hip/hip_bf16.h>

// ---------------------------------------------------------------------------
// AbsoluteTopK SAE: encode (f16 WMMA GEMM, f32 accum) -> radix-select top-64
// by |hidden| per row (deterministic, index-ordered ties) -> sparse decode.
//   B=4096, D=768, H=24576, K=64
// d_out = [recon (B*D f32)] ++ [sparse_hidden (B*H f32)]
// d_ws  = [xh f16 B*D][wh f16 H*D][sIdx i32 B*K][sVal f32 B*K]   (~46 MB)
// ---------------------------------------------------------------------------

#define SAE_B 4096
#define SAE_D 768
#define SAE_H 24576
#define SAE_K 64
#define TOPK_CHUNK (SAE_H / 256)   // 96 contiguous elements per thread

typedef __attribute__((ext_vector_type(16))) _Float16 v16h;
typedef __attribute__((ext_vector_type(8)))  _Float16 v8h;
typedef __attribute__((ext_vector_type(8)))  float    v8f;

union V16U { v8h h[2]; v16h v; };

// ---------------------------- f32 -> f16 convert ---------------------------
__global__ void sae_f32_to_f16(const float* __restrict__ in,
                               _Float16* __restrict__ out, long n) {
  long i = (long)blockIdx.x * blockDim.x + threadIdx.x;
  long stride = (long)gridDim.x * blockDim.x;
  for (; i < n; i += stride) out[i] = (_Float16)in[i];
}

// ------------------------- encoder GEMM (WMMA f16) -------------------------
// hidden[b,h] = sum_d x[b,d]*W[h,d] + b_enc[h]
// Block: 256 threads = 8 waves; block tile 128(M=B rows) x 128(N=H cols).
// Wave tile: 32x64 => acc[2][4] of 16x16 f32 fragments. K step = 32.
__global__ __launch_bounds__(256)
void sae_encode_gemm(const _Float16* __restrict__ xh,
                     const _Float16* __restrict__ wh,
                     const float* __restrict__ b_enc,
                     float* __restrict__ hidden) {
  const int lane = threadIdx.x & 31;
  const int wave = threadIdx.x >> 5;
  const int wm   = wave & 3;          // 0..3 along M
  const int wn   = wave >> 2;         // 0..1 along N
  const int lh   = lane & 15;         // 0..15
  const int lq   = lane >> 4;         // 0 or 1 (half-wave)

  const int bBase = blockIdx.x * 128 + wm * 32;   // rows of x
  const int hBase = blockIdx.y * 128 + wn * 64;   // rows of W / cols of hidden

  v8f acc[2][4];
#pragma unroll
  for (int mi = 0; mi < 2; ++mi)
#pragma unroll
    for (int ni = 0; ni < 4; ++ni)
#pragma unroll
      for (int r = 0; r < 8; ++r) acc[mi][ni][r] = 0.0f;

  for (int d0 = 0; d0 < SAE_D; d0 += 32) {
    // A fragments: 16x32 f16.  lane<16: M=lh, e0..15=K0..15
    //              lane>=16:  M=lh, e0..15=K16..31
    V16U a[2];
#pragma unroll
    for (int mi = 0; mi < 2; ++mi) {
      const _Float16* p = xh + (size_t)(bBase + mi * 16 + lh) * SAE_D + d0 + lq * 16;
      a[mi].h[0] = *(const v8h*)(p);
      a[mi].h[1] = *(const v8h*)(p + 8);
    }
    // B fragments: 32x16 f16.  lane<16: N=lh, e0..15 = K0..15
    //              lane>=16:  N=lh, e0..15 = K16..31  (one contiguous 32B)
    v16h b[4];
#pragma unroll
    for (int ni = 0; ni < 4; ++ni) {
      const _Float16* p = wh + (size_t)(hBase + ni * 16 + lh) * SAE_D + d0 + lq * 16;
      b[ni] = *(const v16h*)(p);
    }
#pragma unroll
    for (int mi = 0; mi < 2; ++mi)
#pragma unroll
      for (int ni = 0; ni < 4; ++ni)
        acc[mi][ni] = __builtin_amdgcn_wmma_f32_16x16x32_f16(
            false, a[mi].v, false, b[ni], (short)0, acc[mi][ni], false, false);
  }

  // Epilogue: C/D layout: lane<16 -> N=lh, VGPR r -> M=r ; lane>=16 -> M=8+r
#pragma unroll
  for (int ni = 0; ni < 4; ++ni) {
    const int col  = hBase + ni * 16 + lh;
    const float bb = b_enc[col];
#pragma unroll
    for (int mi = 0; mi < 2; ++mi) {
      const int rowBase = bBase + mi * 16 + lq * 8;
#pragma unroll
      for (int r = 0; r < 8; ++r)
        hidden[(size_t)(rowBase + r) * SAE_H + col] = acc[mi][ni][r] + bb;
    }
  }
}

// ---------------- per-row abs-top-K radix select (in place) ----------------
__global__ __launch_bounds__(256)
void sae_topk(float* __restrict__ hidden,
              int* __restrict__ sIdx, float* __restrict__ sVal) {
  const int row = blockIdx.x;
  const int tid = threadIdx.x;
  float* rp = hidden + (size_t)row * SAE_H;

  __shared__ unsigned hist[256];
  __shared__ unsigned tGt[256];
  __shared__ unsigned tEq[256];
  __shared__ unsigned s_prefix, s_remaining;

  unsigned prefix = 0, remaining = SAE_K;
#pragma unroll
  for (int shift = 24; shift >= 0; shift -= 8) {
    hist[tid] = 0;
    __syncthreads();
    const int hb = shift + 8;
    for (int j = 0; j < TOPK_CHUNK; ++j) {
      const int i = tid * TOPK_CHUNK + j;
      const unsigned u = __float_as_uint(rp[i]) & 0x7fffffffu;
      const bool cand = (hb >= 32) || ((u >> hb) == (prefix >> hb));
      if (cand) atomicAdd(&hist[(u >> shift) & 0xFFu], 1u);
    }
    __syncthreads();
    if (tid == 0) {
      unsigned cum = 0; int bin = 0;
      for (int bb = 255; bb >= 0; --bb) {
        const unsigned c = hist[bb];
        if (cum + c >= remaining) { bin = bb; break; }
        cum += c;
      }
      s_prefix = prefix | ((unsigned)bin << shift);
      s_remaining = remaining - cum;
    }
    __syncthreads();
    prefix = s_prefix;
    remaining = s_remaining;
    __syncthreads();
  }

  const unsigned T   = prefix;           // K-th largest abs-bit pattern
  const unsigned ngt = SAE_K - remaining; // count strictly greater than T

  // Deterministic ranks: contiguous chunks preserve index order.
  unsigned gt = 0, eq = 0;
  for (int j = 0; j < TOPK_CHUNK; ++j) {
    const unsigned u = __float_as_uint(rp[tid * TOPK_CHUNK + j]) & 0x7fffffffu;
    gt += (u > T);
    eq += (u == T);
  }
  tGt[tid] = gt; tEq[tid] = eq;
  __syncthreads();
  if (tid == 0) {  // serial exclusive scan over 256 entries
    unsigned ag = 0, ae = 0;
    for (int t = 0; t < 256; ++t) {
      const unsigned g = tGt[t], e = tEq[t];
      tGt[t] = ag; tEq[t] = ae;
      ag += g; ae += e;
    }
  }
  __syncthreads();

  unsigned gslot = tGt[tid];   // slots [0, ngt) in index order
  unsigned erank = tEq[tid];   // rank among == T elements, index order
  for (int j = 0; j < TOPK_CHUNK; ++j) {
    const int i = tid * TOPK_CHUNK + j;
    const float v = rp[i];
    const unsigned u = __float_as_uint(v) & 0x7fffffffu;
    if (u > T) {
      sIdx[row * SAE_K + gslot] = i;
      sVal[row * SAE_K + gslot] = v;
      ++gslot;                          // keep rp[i]
    } else if (u == T) {
      if (erank < remaining) {          // first `remaining` ties by index
        const unsigned slot = ngt + erank;
        sIdx[row * SAE_K + slot] = i;
        sVal[row * SAE_K + slot] = v;   // keep rp[i]
      } else {
        rp[i] = 0.0f;
      }
      ++erank;
    } else {
      rp[i] = 0.0f;
    }
  }
}

// ------------------------- sparse decode (tied W) --------------------------
__global__ __launch_bounds__(256)
void sae_decode(const float* __restrict__ W, const float* __restrict__ b_dec,
                const int* __restrict__ sIdx, const float* __restrict__ sVal,
                float* __restrict__ recon) {
  const int row = blockIdx.x;
  const int tid = threadIdx.x;
  __shared__ int   li[SAE_K];
  __shared__ float lv[SAE_K];
  if (tid < SAE_K) {
    li[tid] = sIdx[row * SAE_K + tid];
    lv[tid] = sVal[row * SAE_K + tid];
  }
  __syncthreads();
#pragma unroll
  for (int c = 0; c < SAE_D / 256; ++c) {
    const int d = tid + c * 256;
    float acc = b_dec[d];
#pragma unroll 8
    for (int j = 0; j < SAE_K; ++j)
      acc = fmaf(lv[j], W[(size_t)li[j] * SAE_D + d], acc);
    recon[(size_t)row * SAE_D + d] = acc;
  }
}

// ------------------------------- launcher ----------------------------------
extern "C" void kernel_launch(void* const* d_in, const int* in_sizes, int n_in,
                              void* d_out, int out_size, void* d_ws, size_t ws_size,
                              hipStream_t stream) {
  const float* x     = (const float*)d_in[0];  // [B, D]
  const float* W     = (const float*)d_in[1];  // [H, D]
  const float* b_enc = (const float*)d_in[2];  // [H]
  const float* b_dec = (const float*)d_in[3];  // [D]

  float* recon  = (float*)d_out;                     // [B, D]
  float* hidden = recon + (size_t)SAE_B * SAE_D;     // [B, H] (sparse output)

  char* ws = (char*)d_ws;
  _Float16* xh = (_Float16*)ws;                              // B*D halves
  _Float16* wh = xh + (size_t)SAE_B * SAE_D;                 // H*D halves
  int*   sIdx  = (int*)(wh + (size_t)SAE_H * SAE_D);         // B*K ints
  float* sVal  = (float*)(sIdx + (size_t)SAE_B * SAE_K);     // B*K floats

  sae_f32_to_f16<<<2048, 256, 0, stream>>>(x, xh, (long)SAE_B * SAE_D);
  sae_f32_to_f16<<<8192, 256, 0, stream>>>(W, wh, (long)SAE_H * SAE_D);

  dim3 grid(SAE_B / 128, SAE_H / 128);  // 32 x 192
  sae_encode_gemm<<<grid, 256, 0, stream>>>(xh, wh, b_enc, hidden);

  sae_topk<<<SAE_B, 256, 0, stream>>>(hidden, sIdx, sVal);

  sae_decode<<<SAE_B, 256, 0, stream>>>(W, b_dec, sIdx, sVal, recon);
}
